// DecoderRNN_84997402788354
// MI455X (gfx1250) — matrix operand order repaired and checked by reading, not compiled, not run on hardware
//
#include <hip/hip_runtime.h>
#include <hip/hip_bf16.h>

// ---------------- problem constants ----------------
#define TT   127        // time steps (SEQ-1)
#define BB   32         // batch
#define HH   512        // hidden = emb
#define VV   32000      // vocab
#define G3   1536       // 3*HID
#define RR   (TT*BB)    // 4064 rows

typedef __bf16 bf16;
typedef __attribute__((ext_vector_type(8)))  __bf16 bf16x8;
typedef __attribute__((ext_vector_type(16))) __bf16 v16bf;
typedef __attribute__((ext_vector_type(8)))  float  v8f;

// ---------------- WMMA fragment helpers (CDNA5 16x16x32 bf16) ----------------
// A tile 16x32 (MxK), source row-major with leading dim ld (bf16).
// Lanes 0-15: M=lane, K = {0..7, 16..23}; lanes 16-31: M=lane-16, K = {8..15, 24..31}.
__device__ inline v16bf load_a_frag(const bf16* base, int ld, int lane) {
  int row = lane & 15;
  int ko  = (lane >> 4) << 3;               // 0 or 8
  const bf16* p = base + row * ld + ko;
  union { v16bf v; bf16x8 h[2]; } u;
  u.h[0] = *(const bf16x8*)(p);
  u.h[1] = *(const bf16x8*)(p + 16);
  return u.v;
}

// Same A layout but source is f32 (LDS hidden state) — convert on load.
__device__ inline v16bf a_frag_from_f32(const float* base, int ld, int lane, int k0) {
  int row = lane & 15;
  int ko  = (lane >> 4) << 3;
  const float* p = base + row * ld + k0 + ko;
  v16bf f;
#pragma unroll
  for (int i = 0; i < 8; ++i) {
    f[i]     = (bf16)p[i];
    f[i + 8] = (bf16)p[i + 16];
  }
  return f;
}

// B tile 32x16 (KxN). Source: W row-major (N x K) with leading dim ld, i.e. B = W^T.
// Lanes 0-15: N=lane, K = {0..7, 16..23}; lanes 16-31: N=lane-16, K = {8..15, 24..31}.
__device__ inline v16bf load_b_frag(const bf16* W, int ld, int n0, int k0, int lane) {
  int col = lane & 15;
  int ko  = (lane >> 4) << 3;               // 0 or 8
  const bf16* p = W + (size_t)(n0 + col) * ld + k0 + ko;
  union { v16bf v; bf16x8 h[2]; } u;
  u.h[0] = *(const bf16x8*)(p);
  u.h[1] = *(const bf16x8*)(p + 16);
  return u.v;
}

__device__ inline v8f wmma_bf16(v16bf a, v16bf b, v8f c) {
  return __builtin_amdgcn_wmma_f32_16x16x32_bf16(false, a, false, b, (short)0, c,
                                                 false, false);
}

__device__ inline v8f zero8() { v8f z = {0.f,0.f,0.f,0.f,0.f,0.f,0.f,0.f}; return z; }

// ---------------- kernel 1: f32 -> bf16 convert ----------------
__global__ void cvt_bf16(const float* __restrict__ in, bf16* __restrict__ out, size_t n) {
  size_t i  = (size_t)blockIdx.x * blockDim.x + threadIdx.x;
  size_t st = (size_t)gridDim.x * blockDim.x;
  for (; i < n; i += st) out[i] = (bf16)in[i];
}

// ---------------- kernel 2: embedding gather -> bf16, layout [t][b][k] ----------------
__global__ void embed_gather(const int* __restrict__ target, const float* __restrict__ emb,
                             bf16* __restrict__ Xb) {
  size_t i = (size_t)blockIdx.x * blockDim.x + threadIdx.x;
  if (i >= (size_t)RR * HH) return;
  int k = (int)(i & (HH - 1));
  int r = (int)(i >> 9);            // row = t*32 + b
  int t = r >> 5;
  int b = r & 31;
  int tok = target[b * 128 + t];    // target is (BATCH=32, SEQ=128); use t in [0,126]
  Xb[i] = (bf16)emb[(size_t)tok * HH + k];
}

// ---------------- kernel 3: gi = Xb @ W_ih^T + b_ih  (RR x G3, K=HH) ----------------
__global__ void __launch_bounds__(256)
gemm_gi(const bf16* __restrict__ Xb, const bf16* __restrict__ Wb,
        const float* __restrict__ bih, float* __restrict__ gi) {
  const int NT = G3 / 16;                                   // 96 N-tiles
  int wave = blockIdx.x * (blockDim.x >> 5) + (threadIdx.x >> 5);
  int lane = threadIdx.x & 31;
  if (wave >= (RR / 16) * NT) return;                       // wave-uniform
  int mt = wave / NT, nt = wave % NT;
  const bf16* Arow = Xb + (size_t)mt * 16 * HH;
  v8f acc = zero8();
  for (int kk = 0; kk < HH; kk += 32) {
    v16bf a = load_a_frag(Arow + kk, HH, lane);
    v16bf b = load_b_frag(Wb, HH, nt * 16, kk, lane);
    acc = wmma_bf16(a, b, acc);
  }
  int n  = nt * 16 + (lane & 15);
  int m0 = mt * 16 + ((lane >> 4) << 3);
  float bias = bih[n];
#pragma unroll
  for (int j = 0; j < 8; ++j)
    gi[(size_t)(m0 + j) * G3 + n] = acc[j] + bias;
}

// ---------------- kernel 4: sequential GRU, single WGP (32 wave32s) ----------------
__global__ void __launch_bounds__(1024)
gru_seq(const bf16* __restrict__ Whh, const float* __restrict__ bhh,
        const float* __restrict__ gi, const float* __restrict__ h0,
        bf16* __restrict__ hs) {
  __shared__ __align__(16) float hL[BB * HH];               // 64 KB hidden state
  int tid = threadIdx.x;
  for (int i = tid; i < BB * HH; i += 1024) hL[i] = h0[i];
  __syncthreads();

  int wave = tid >> 5, lane = tid & 31;
  int col = lane & 15, grp = lane >> 4;

  for (int t = 0; t < TT; ++t) {
    const float* giT = gi + (size_t)t * BB * G3;
    float hnew[2][8];
#pragma unroll
    for (int jb = 0; jb < 2; ++jb) {
      int job = wave + jb * 32;          // 64 jobs: 2 M-tiles x 32 hidden-tiles
      int mt  = job >> 5;                // 0..1
      int ht  = job & 31;                // hidden tile 0..31
      v8f ar = zero8(), az = zero8(), an = zero8();
      const float* Abase = hL + mt * 16 * HH;
      for (int kk = 0; kk < HH; kk += 32) {
        v16bf a  = a_frag_from_f32(Abase, HH, lane, kk);
        v16bf br = load_b_frag(Whh, HH, 0 * HH + ht * 16, kk, lane);
        v16bf bz = load_b_frag(Whh, HH, 1 * HH + ht * 16, kk, lane);
        v16bf bn = load_b_frag(Whh, HH, 2 * HH + ht * 16, kk, lane);
        ar = wmma_bf16(a, br, ar);
        az = wmma_bf16(a, bz, az);
        an = wmma_bf16(a, bn, an);
      }
      int n  = ht * 16 + col;
      int m0 = mt * 16 + grp * 8;
      float bhr = bhh[n], bhz = bhh[HH + n], bhn = bhh[2 * HH + n];
#pragma unroll
      for (int j = 0; j < 8; ++j) {
        int m    = m0 + j;
        float ir  = giT[(size_t)m * G3 + n];
        float iz  = giT[(size_t)m * G3 + HH + n];
        float inn = giT[(size_t)m * G3 + 2 * HH + n];
        float hr = ar[j] + bhr, hz = az[j] + bhz, hn = an[j] + bhn;
        float r  = 1.f / (1.f + __expf(-(ir + hr)));
        float z  = 1.f / (1.f + __expf(-(iz + hz)));
        float nn = tanhf(inn + r * hn);
        float hp = hL[m * HH + n];
        hnew[jb][j] = (1.f - z) * nn + z * hp;
      }
    }
    __syncthreads();                     // all reads of old h done
#pragma unroll
    for (int jb = 0; jb < 2; ++jb) {
      int job = wave + jb * 32;
      int mt = job >> 5, ht = job & 31;
      int n  = ht * 16 + col;
      int m0 = mt * 16 + grp * 8;
#pragma unroll
      for (int j = 0; j < 8; ++j) {
        int m = m0 + j;
        hL[m * HH + n] = hnew[jb][j];
        hs[((size_t)t * BB + m) * HH + n] = (bf16)hnew[jb][j];
      }
    }
    __syncthreads();
  }
}

// ---------------- kernel 5: logits = hs @ out_W^T + b, fused online logsumexp ----------
__global__ void __launch_bounds__(256)
out_gemm(const bf16* __restrict__ hs, const bf16* __restrict__ Wo,
         const float* __restrict__ bo, float* __restrict__ out,
         float* __restrict__ lse) {
  __shared__ __align__(16) bf16 AL[16 * HH];                // 16 KB: 16 (t,b) rows
  __shared__ float Rmx[16 * 128];
  __shared__ float Rsm[16 * 128];
  int tid = threadIdx.x, wave = tid >> 5, lane = tid & 31;
  int col = lane & 15, grp = lane >> 4;
  int r0 = blockIdx.x * 16;

  for (int i = tid; i < 16 * HH; i += 256) AL[i] = hs[(size_t)r0 * HH + i];
  __syncthreads();

  float mx[8], sm[8];
#pragma unroll
  for (int j = 0; j < 8; ++j) { mx[j] = -3.4e38f; sm[j] = 0.f; }

  for (int vt = wave; vt < VV / 16; vt += 8) {
    int v0 = vt * 16;
    // prefetch next B tile this wave will touch
    if (v0 + 128 < VV)
      __builtin_prefetch(Wo + (size_t)(v0 + 128 + col) * HH, 0, 1);
    v8f acc = zero8();
    for (int kk = 0; kk < HH; kk += 32) {
      v16bf a = load_a_frag(AL + kk, HH, lane);
      v16bf b = load_b_frag(Wo, HH, v0, kk, lane);
      acc = wmma_bf16(a, b, acc);
    }
    int v = v0 + col;
    float bias = bo[v];
    int m0 = grp * 8;
#pragma unroll
    for (int j = 0; j < 8; ++j) {
      float val = acc[j] + bias;
      __builtin_nontemporal_store(val, &out[((size_t)(r0 + m0 + j)) * VV + v]);
      if (val > mx[j]) { sm[j] = sm[j] * __expf(mx[j] - val) + 1.f; mx[j] = val; }
      else             { sm[j] += __expf(val - mx[j]); }
    }
  }

  int slot = wave * 16 + col;           // 0..127
#pragma unroll
  for (int j = 0; j < 8; ++j) {
    int row = grp * 8 + j;
    Rmx[row * 128 + slot] = mx[j];
    Rsm[row * 128 + slot] = sm[j];
  }
  __syncthreads();
  if (tid < 16) {
    float M = -3.4e38f, S = 0.f;
    for (int i = 0; i < 128; ++i) {
      float m2 = Rmx[tid * 128 + i], s2 = Rsm[tid * 128 + i];
      if (m2 > M) { S = S * __expf(M - m2) + s2; M = m2; }
      else        { S += s2 * __expf(m2 - M); }
    }
    lse[r0 + tid] = M + __logf(S);
  }
}

// ---------------- kernel 6: logp = logits - lse[row] ----------------
__global__ void __launch_bounds__(256)
fixup(float* __restrict__ out, const float* __restrict__ lse) {
  int row = blockIdx.y;
  int c   = blockIdx.x * blockDim.x + threadIdx.x;
  if (c >= VV) return;
  size_t idx = (size_t)row * VV + c;
  float v = out[idx];
  out[idx] = v - lse[row];
}

// ---------------- workspace layout (bytes, 256-aligned) ----------------
#define OFF_WIHB 0u
#define OFF_WHHB 1572864u
#define OFF_WOB  3145728u
#define OFF_XB   35913728u
#define OFF_GI   40075264u
#define OFF_HS   65044480u
#define OFF_LSE  69206016u

extern "C" void kernel_launch(void* const* d_in, const int* in_sizes, int n_in,
                              void* d_out, int out_size, void* d_ws, size_t ws_size,
                              hipStream_t stream) {
  const int*   target = (const int*)  d_in[0];
  const float* enc    = (const float*)d_in[1];   // (1, B, H)
  const float* emb    = (const float*)d_in[2];
  const float* Wih    = (const float*)d_in[3];
  const float* Whh    = (const float*)d_in[4];
  const float* bih    = (const float*)d_in[5];
  const float* bhh    = (const float*)d_in[6];
  const float* Wo     = (const float*)d_in[7];
  const float* bo     = (const float*)d_in[8];
  float* out = (float*)d_out;

  char* ws = (char*)d_ws;
  bf16*  Wihb = (bf16*)(ws + OFF_WIHB);
  bf16*  Whhb = (bf16*)(ws + OFF_WHHB);
  bf16*  Wob  = (bf16*)(ws + OFF_WOB);
  bf16*  Xb   = (bf16*)(ws + OFF_XB);
  float* gi   = (float*)(ws + OFF_GI);
  bf16*  hsb  = (bf16*)(ws + OFF_HS);
  float* lse  = (float*)(ws + OFF_LSE);

  // 1. weight conversions to bf16
  cvt_bf16<<<2048, 256, 0, stream>>>(Wih, Wihb, (size_t)G3 * HH);
  cvt_bf16<<<2048, 256, 0, stream>>>(Whh, Whhb, (size_t)G3 * HH);
  cvt_bf16<<<4096, 256, 0, stream>>>(Wo,  Wob,  (size_t)VV * HH);

  // 2. embedding gather -> Xb [t][b][k] bf16
  embed_gather<<<(RR * HH) / 256, 256, 0, stream>>>(target, emb, Xb);

  // 3. gi = Xb @ W_ih^T + b_ih  (all timesteps in parallel)
  //    waves = 254 M-tiles * 96 N-tiles = 24384; 8 waves/block -> 3048 blocks
  gemm_gi<<<3048, 256, 0, stream>>>(Xb, Wihb, bih, gi);

  // 4. sequential GRU on one WGP
  gru_seq<<<1, 1024, 0, stream>>>(Whhb, bhh, gi, enc, hsb);

  // 5. output projection + online logsumexp (254 blocks, 16 rows each)
  out_gemm<<<RR / 16, 256, 0, stream>>>(hsb, Wob, bo, out, lse);

  // 6. logp = logits - lse
  fixup<<<dim3(VV / 256, RR), 256, 0, stream>>>(out, lse);
}
